// Vit_Attention_9131100471522
// MI455X (gfx1250) — matrix operand order repaired and checked by reading, not compiled, not run on hardware
//
#include <hip/hip_runtime.h>

// ---------------------------------------------------------------------------
// ViT attention for MI455X (gfx1250, wave32, WMMA).
// f16 compute / f32 accumulate via v_wmma_f32_16x16x32_f16.
// Double-buffered LDS pipeline; per-wave 32x64 output tile (8 WMMA / chunk).
// ---------------------------------------------------------------------------

typedef _Float16 v16h __attribute__((ext_vector_type(16)));
typedef _Float16 v8h  __attribute__((ext_vector_type(8)));
typedef float    v8f  __attribute__((ext_vector_type(8)));

#define EMBED  768
#define NHEADS 12
#define HDIM   64
#define BATCH  4
#define SEQ    2048

#define BM 256    // block tile M (8 waves x 32 rows)
#define BN 64     // block tile N
#define BK 32     // K chunk (matches 16x16x32 WMMA K)
#define LDSPAD 8  // half-element row padding (16B) to stagger LDS banks
#define BKP (BK + LDSPAD)

static __device__ __forceinline__ v16h cat8(v8h a, v8h b) {
    return __builtin_shufflevector(a, b, 0,1,2,3,4,5,6,7,8,9,10,11,12,13,14,15);
}

static __device__ __forceinline__ v8h cvt8(float4 f0, float4 f1) {
    v8h h;
    h[0]=(_Float16)f0.x; h[1]=(_Float16)f0.y; h[2]=(_Float16)f0.z; h[3]=(_Float16)f0.w;
    h[4]=(_Float16)f1.x; h[5]=(_Float16)f1.y; h[6]=(_Float16)f1.z; h[7]=(_Float16)f1.w;
    return h;
}

// Stage one K-chunk: A tile BMx32 (one 32-float row per thread) and B tile
// 64x32 into LDS as f16, B always stored N-major / K-contiguous.
static __device__ __forceinline__
void stage_tiles(const float* __restrict__ A, long long lda, long long mbase,
                 const float* __restrict__ Bm, long long ldb, long long nbase,
                 int b_is_kn, int kb, int tid,
                 _Float16 (*__restrict__ As)[BKP],
                 _Float16 (*__restrict__ Bs)[BKP])
{
    // ---- A: thread `tid` copies row `tid`, 32 f32 -> 32 f16 ----
    const float4* srcA = (const float4*)(A + (mbase + tid) * lda + kb);
#pragma unroll
    for (int j = 0; j < 4; ++j) {
        *(v8h*)&As[tid][j * 8] = cvt8(srcA[2 * j], srcA[2 * j + 1]);
    }
    // ---- B tile -> Bs[n][k] ----
    if (!b_is_kn) {
        // B[N][K]: rows are K-contiguous, direct copy (4 threads per row)
        const int n  = tid >> 2;
        const int kp = (tid & 3) * 8;
        const float4* src = (const float4*)(Bm + (nbase + n) * ldb + kb + kp);
        *(v8h*)&Bs[n][kp] = cvt8(src[0], src[1]);
    } else {
        // B[K][N]: read a K-row (N-contiguous), transpose into LDS
        const int k  = tid >> 3;
        const int np = (tid & 7) * 8;
        const float4* src = (const float4*)(Bm + (long long)(kb + k) * ldb + nbase + np);
        float4 f0 = src[0], f1 = src[1];
        Bs[np + 0][k] = (_Float16)f0.x;
        Bs[np + 1][k] = (_Float16)f0.y;
        Bs[np + 2][k] = (_Float16)f0.z;
        Bs[np + 3][k] = (_Float16)f0.w;
        Bs[np + 4][k] = (_Float16)f1.x;
        Bs[np + 5][k] = (_Float16)f1.y;
        Bs[np + 6][k] = (_Float16)f1.z;
        Bs[np + 7][k] = (_Float16)f1.w;
    }
}

// Generic batched GEMM:  C = alpha * (A x B') + bias
//   A:  [M x K], row stride lda (fp32)
//   B:  b_is_kn==0 : B[N][K] row stride ldb  ("NT": C += A . B^T)
//       b_is_kn==1 : B[K][N] row stride ldb  ("NN": C += A . B)
//   batch offsets: z = blockIdx.z; zb = z / zdiv; zh = z % zdiv;
__global__ __launch_bounds__(256)
void wmma_gemm_f32(const float* __restrict__ A,  long long lda, long long a_sb, long long a_sh,
                   const float* __restrict__ Bm, long long ldb, long long b_sb, long long b_sh,
                   int b_is_kn,
                   float* __restrict__ C,        long long ldc, long long c_sb, long long c_sh,
                   const float* __restrict__ bias, float alpha, int K, int zdiv)
{
    __shared__ __align__(32) _Float16 As[2][BM][BKP];   // 2 x 20 KB
    __shared__ __align__(32) _Float16 Bs[2][BN][BKP];   // 2 x  5 KB

    const int z  = blockIdx.z;
    const int zb = z / zdiv;
    const int zh = z % zdiv;
    A  += (long long)zb * a_sb + (long long)zh * a_sh;
    Bm += (long long)zb * b_sb + (long long)zh * b_sh;
    C  += (long long)zb * c_sb + (long long)zh * c_sh;

    const int tid  = threadIdx.x;
    const int lane = tid & 31;
    const int wave = tid >> 5;
    const long long mbase = (long long)blockIdx.x * BM;
    const long long nbase = (long long)blockIdx.y * BN;

    v8f acc0[4] = {};   // rows [wave*32 .. +16)
    v8f acc1[4] = {};   // rows [wave*32+16 .. +32)

    // prologue: fill buffer 0
    stage_tiles(A, lda, mbase, Bm, ldb, nbase, b_is_kn, 0, tid, As[0], Bs[0]);
    __syncthreads();

    int buf = 0;
    for (int kb = 0; kb < K; kb += BK) {
        // stage next chunk into the other buffer while this one computes
        if (kb + BK < K) {
            stage_tiles(A, lda, mbase, Bm, ldb, nbase, b_is_kn,
                        kb + BK, tid, As[buf ^ 1], Bs[buf ^ 1]);
            if (kb + 2 * BK < K)   // prefetch chunk after next (global_prefetch_b8)
                __builtin_prefetch(A + (mbase + tid) * lda + (kb + 2 * BK), 0, 3);
        }

        // ---- WMMA: wave computes 32 rows x 64 cols ----
        {
            // A fragment, 16x32 f16 per ISA layout: lane -> M = lane&15,
            // lanes 0-15 hold K {0..7,16..23}, lanes 16-31 hold K {8..15,24..31}
            const int frow = wave * 32 + (lane & 15);
            const int ksel = (lane >> 4) * 8;
            v16h a0 = cat8(*(const v8h*)&As[buf][frow][ksel],
                           *(const v8h*)&As[buf][frow][ksel + 16]);
            v16h a1 = cat8(*(const v8h*)&As[buf][frow + 16][ksel],
                           *(const v8h*)&As[buf][frow + 16][ksel + 16]);
            // B fragment, 32x16 f16: lane -> N = lane&15,
            // lanes 0-15 hold K {0..7,16..23}, lanes 16-31 hold K {8..15,24..31}
            const int bcol = lane & 15;
            const int koff = (lane >> 4) * 8;
#pragma unroll
            for (int t = 0; t < 4; ++t) {
                v16h bf = cat8(*(const v8h*)&Bs[buf][t * 16 + bcol][koff],
                               *(const v8h*)&Bs[buf][t * 16 + bcol][koff + 16]);
                acc0[t] = __builtin_amdgcn_wmma_f32_16x16x32_f16(
                              false, a0, false, bf, (short)0, acc0[t], false, false);
                acc1[t] = __builtin_amdgcn_wmma_f32_16x16x32_f16(
                              false, a1, false, bf, (short)0, acc1[t], false, false);
            }
        }
        __syncthreads();
        buf ^= 1;
    }

    // ---- epilogue: C/D layout -> VGPR r holds M = r + (lane>=16 ? 8:0), N = lane&15
    const int mrow  = wave * 32 + ((lane >> 4) << 3);
    const int ncol0 = lane & 15;
#pragma unroll
    for (int t = 0; t < 4; ++t) {
        const long long n = nbase + t * 16 + ncol0;
        const float bv = bias ? bias[n] : 0.0f;
#pragma unroll
        for (int r = 0; r < 8; ++r) {
            const long long m0 = mbase + mrow + r;
            C[m0 * ldc + n]        = acc0[t][r] * alpha + bv;
            C[(m0 + 16) * ldc + n] = acc1[t][r] * alpha + bv;
        }
    }
}

// Row softmax over 2048 columns; one 256-thread block per row; values kept in
// registers so the 805MB matrix is read once and written once.
__global__ __launch_bounds__(256)
void softmax2048(float* __restrict__ attn)
{
    __shared__ float red[256];
    const long long row = blockIdx.x;
    float4* p4 = (float4*)(attn + row * (long long)SEQ);
    const int tid = threadIdx.x;

    float4 a = p4[tid];
    float4 b = p4[tid + 256];

    float m = fmaxf(fmaxf(fmaxf(a.x, a.y), fmaxf(a.z, a.w)),
                    fmaxf(fmaxf(b.x, b.y), fmaxf(b.z, b.w)));
    red[tid] = m; __syncthreads();
    for (int s = 128; s > 0; s >>= 1) {
        if (tid < s) red[tid] = fmaxf(red[tid], red[tid + s]);
        __syncthreads();
    }
    m = red[0]; __syncthreads();

    a.x = __expf(a.x - m); a.y = __expf(a.y - m);
    a.z = __expf(a.z - m); a.w = __expf(a.w - m);
    b.x = __expf(b.x - m); b.y = __expf(b.y - m);
    b.z = __expf(b.z - m); b.w = __expf(b.w - m);

    red[tid] = (a.x + a.y + a.z + a.w) + (b.x + b.y + b.z + b.w);
    __syncthreads();
    for (int s = 128; s > 0; s >>= 1) {
        if (tid < s) red[tid] += red[tid + s];
        __syncthreads();
    }
    const float inv = 1.0f / red[0];

    a.x *= inv; a.y *= inv; a.z *= inv; a.w *= inv;
    b.x *= inv; b.y *= inv; b.z *= inv; b.w *= inv;
    p4[tid] = a; p4[tid + 256] = b;
}

extern "C" void kernel_launch(void* const* d_in, const int* in_sizes, int n_in,
                              void* d_out, int out_size, void* d_ws, size_t ws_size,
                              hipStream_t stream)
{
    (void)in_sizes; (void)n_in; (void)out_size; (void)ws_size;

    const float* hidden = (const float*)d_in[0];
    const float* Wq = (const float*)d_in[1];
    const float* bq = (const float*)d_in[2];
    const float* Wk = (const float*)d_in[3];
    const float* bk = (const float*)d_in[4];
    const float* Wv = (const float*)d_in[5];
    const float* bv = (const float*)d_in[6];
    const float* Wo = (const float*)d_in[7];
    const float* bo = (const float*)d_in[8];

    float* out      = (float*)d_out;
    float* attn_out = out;                                       // [4,2048,768]
    float* attn_w   = out + (long long)BATCH * SEQ * EMBED;      // [4,12,2048,2048]

    const long long NE = (long long)BATCH * SEQ * EMBED;         // 6,291,456
    float* Q   = (float*)d_ws;
    float* Kp  = Q  + NE;
    float* V   = Kp + NE;
    float* CTX = V  + NE;                                        // ws: ~100 MB

    const long long SS  = (long long)SEQ * SEQ;                  // 4,194,304
    const long long BSE = (long long)SEQ * EMBED;

    dim3 blk(256);
    dim3 gProj(BATCH * SEQ / BM, EMBED / BN, 1);                 // 32 x 12

    // Q/K/V projections: x @ W^T + b   (NT)
    wmma_gemm_f32<<<gProj, blk, 0, stream>>>(hidden, EMBED, 0, 0,
                                             Wq, EMBED, 0, 0, 0,
                                             Q,  EMBED, 0, 0, bq, 1.0f, EMBED, 1);
    wmma_gemm_f32<<<gProj, blk, 0, stream>>>(hidden, EMBED, 0, 0,
                                             Wk, EMBED, 0, 0, 0,
                                             Kp, EMBED, 0, 0, bk, 1.0f, EMBED, 1);
    wmma_gemm_f32<<<gProj, blk, 0, stream>>>(hidden, EMBED, 0, 0,
                                             Wv, EMBED, 0, 0, 0,
                                             V,  EMBED, 0, 0, bv, 1.0f, EMBED, 1);

    // Scores: per (b,h)  S = scale * Q_h @ K_h^T   (NT, K=64)
    const float scale = 0.125f; // HEAD_DIM^-0.5
    dim3 gScore(SEQ / BM, SEQ / BN, BATCH * NHEADS);             // 8 x 32 x 48
    wmma_gemm_f32<<<gScore, blk, 0, stream>>>(
        Q,      EMBED, BSE, HDIM,                 // A: + b*S*E + h*64
        Kp,     EMBED, BSE, HDIM, 0,              // B: + b*S*E + h*64
        attn_w, SEQ, (long long)NHEADS * SS, SS,  // C: + (b*12+h)*S*S
        nullptr, scale, HDIM, NHEADS);

    // Softmax over last dim (in place in d_out attn_weights region)
    softmax2048<<<dim3(BATCH * NHEADS * SEQ), blk, 0, stream>>>(attn_w);

    // Context: per (b,h)  ctx = P @ V_h   (NN, K=2048)
    dim3 gPV(SEQ / BM, HDIM / BN, BATCH * NHEADS);               // 8 x 1 x 48
    wmma_gemm_f32<<<gPV, blk, 0, stream>>>(
        attn_w, SEQ, (long long)NHEADS * SS, SS,
        V,      EMBED, BSE, HDIM, 1,
        CTX,    EMBED, BSE, HDIM,
        nullptr, 1.0f, SEQ, NHEADS);

    // Output projection: ctx @ Wo^T + bo   (NT)
    wmma_gemm_f32<<<gProj, blk, 0, stream>>>(CTX, EMBED, 0, 0,
                                             Wo, EMBED, 0, 0, 0,
                                             attn_out, EMBED, 0, 0, bo, 1.0f, EMBED, 1);
}
